// CONAD_Base_19567871001158
// MI455X (gfx1250) — matrix operand order repaired and checked by reading, not compiled, not run on hardware
//
#include <hip/hip_runtime.h>
#include <hip/hip_bf16.h>

typedef __attribute__((ext_vector_type(16))) __bf16 v16bf;
typedef __attribute__((ext_vector_type(8)))  __bf16 v8bf;
typedef __attribute__((ext_vector_type(8)))  float  v8f;

#define NNODES 10000
#define INDIM  256
#define HIDDIM 64

#define USE_ASYNC_LDS 1

// ---------------------------------------------------------------------------
// 16-byte global->LDS copy. Async path: per-lane global_load_async_to_lds_b128
// (GV mode: vdst = LDS byte address, vaddr = 64-bit global address). The low
// 32 bits of a flat shared-memory pointer ARE the LDS byte offset (ISA 10.2:
// LDS aperture maps addr[31:0] -> LDS_ADDR). Out-of-bounds lanes zero-fill
// with a normal ds store (covered by DScnt / barrier).
// ---------------------------------------------------------------------------
__device__ __forceinline__ void cp16_to_lds(void* lds, const void* g, bool pred) {
#if USE_ASYNC_LDS
  if (pred) {
    unsigned l = (unsigned)(uintptr_t)lds;
    asm volatile("global_load_async_to_lds_b128 %0, %1, off"
                 :: "v"(l), "v"(g) : "memory");
  } else {
    *(uint4*)lds = make_uint4(0u, 0u, 0u, 0u);
  }
#else
  uint4 v = make_uint4(0u, 0u, 0u, 0u);
  if (pred) v = *(const uint4*)g;
  *(uint4*)lds = v;
#endif
}

__device__ __forceinline__ void wait_lds_ready() {
#if USE_ASYNC_LDS
  asm volatile("s_wait_asynccnt 0x0" ::: "memory");
#endif
}

// ---------------------------------------------------------------------------
// Elementwise / graph kernels
// ---------------------------------------------------------------------------
__global__ void k_zero(float* __restrict__ p, int n) {
  int i = blockIdx.x * blockDim.x + threadIdx.x;
  if (i < n) p[i] = 0.0f;
}

__global__ void k_cvt_bf16(__bf16* __restrict__ dst, const float* __restrict__ src, int n) {
  int i = blockIdx.x * blockDim.x + threadIdx.x;
  if (i < n) dst[i] = (__bf16)src[i];
}

__global__ void k_degree(const int* __restrict__ src, const int* __restrict__ dst,
                         int E, float* __restrict__ deg) {
  int e = blockIdx.x * blockDim.x + threadIdx.x;
  if (e < E) {
    if (src[e] != dst[e]) atomicAdd(&deg[dst[e]], 1.0f);
  }
}

__global__ void k_dinv(const float* __restrict__ deg, float* __restrict__ dinv, int n) {
  int i = blockIdx.x * blockDim.x + threadIdx.x;
  if (i < n) dinv[i] = rsqrtf(deg[i] + 1.0f);  // +1 = appended self loop
}

__global__ void k_norm(const int* __restrict__ src, const int* __restrict__ dst,
                       const float* __restrict__ dinv, float* __restrict__ norm, int E) {
  int e = blockIdx.x * blockDim.x + threadIdx.x;
  if (e < E) {
    int s = src[e], d = dst[e];
    norm[e] = (s != d) ? dinv[s] * dinv[d] : 0.0f;  // dropped self loops: w=0
  }
}

// out[i][f] = xw[i][f] * dinv[i]^2 + bias[f]   (self-loop term + bias; full init)
__global__ void k_selfinit(float* __restrict__ out, const float* __restrict__ xw,
                           const float* __restrict__ dinv, const float* __restrict__ bias,
                           int n, int F) {
  long long idx = (long long)blockIdx.x * blockDim.x + threadIdx.x;
  if (idx < (long long)n * F) {
    int i = (int)(idx / F);
    int f = (int)(idx - (long long)i * F);
    float d = dinv[i];
    out[idx] = xw[idx] * d * d + bias[f];
  }
}

// out[dst[e]][f] += xw[src[e]][f] * norm[e]
__global__ void k_scatter(float* __restrict__ out, const float* __restrict__ xw,
                          const int* __restrict__ src, const int* __restrict__ dst,
                          const float* __restrict__ norm, int E, int F) {
  long long idx = (long long)blockIdx.x * blockDim.x + threadIdx.x;
  if (idx < (long long)E * F) {
    int e = (int)(idx / F);
    int f = (int)(idx - (long long)e * F);
    float w = norm[e];
    if (w != 0.0f) {
      atomicAdd(&out[(long long)dst[e] * F + f], xw[(long long)src[e] * F + f] * w);
    }
  }
}

__global__ void k_relu(float* __restrict__ p, int n) {
  int i = blockIdx.x * blockDim.x + threadIdx.x;
  if (i < n) p[i] = fmaxf(p[i], 0.0f);
}

// ---------------------------------------------------------------------------
// bf16 WMMA GEMM (bf16 in, f32 out):
//   TRANSB=0:  C[M,Nf] = A[M,K] @ B[K,Nf]
//   TRANSB=1:  C[M,Nf] = A[M,K] @ B[Nf,K]^T
// 256 threads = 8 wave32 waves; tile BM=64, BN=64; wave (wm,wn) in 4x2 grid
// owns a 16x32 slab -> 2 WMMAs per K-step sharing one A fragment.
// K must be a multiple of 32 (256 or 64 here). M, Nf edge-guarded.
// LDS: As[64][32] bf16 row-major; BsT[64][32] bf16 = B^T so both fragment
// reads are contiguous 16B chunks (k = 16*half + s, same order in A and B).
// ---------------------------------------------------------------------------
template <bool TRANSB>
__global__ void __launch_bounds__(256)
gemm_wmma_bf16(const __bf16* __restrict__ A, const __bf16* __restrict__ B,
               float* __restrict__ C, int M, int K, int Nf) {
  constexpr int BM = 64, BN = 64, BK = 32;
  __shared__ __bf16 As[BM][BK];    // 4 KB (rows = 64 B)
  __shared__ __bf16 BsT[BN][BK];   // 4 KB (BsT[n][k])

  const int tid  = threadIdx.x;
  const int lane = tid & 31;
  const int wave = tid >> 5;
  const int wm = wave >> 1;            // 0..3
  const int wn = wave & 1;             // 0..1
  const int m0 = blockIdx.y * BM;
  const int n0 = blockIdx.x * BN;
  const int ml   = lane & 15;
  const int half = lane >> 4;

  const int ar = tid >> 2;             // staging: row 0..63
  const int ac = tid & 3;              // staging: 16B chunk 0..3

  v8f acc0 = {0.f, 0.f, 0.f, 0.f, 0.f, 0.f, 0.f, 0.f};
  v8f acc1 = {0.f, 0.f, 0.f, 0.f, 0.f, 0.f, 0.f, 0.f};

  for (int k0 = 0; k0 < K; k0 += BK) {
    // prefetch next A K-tile (global_prefetch_b8)
    if (k0 + BK < K) {
      int pr = m0 + ar;
      if (pr < M) __builtin_prefetch(A + (size_t)pr * K + k0 + BK + ac * 8, 0, 0);
    }

    // -- stage A tile 64x32 bf16: 256 async b128 copies --
    {
      int gr = m0 + ar;
      cp16_to_lds(&As[ar][ac * 8], A + (size_t)gr * K + k0 + ac * 8, gr < M);
    }

    // -- stage B tile as BsT[n][k] --
    if (TRANSB) {
      // B[Nf][K] row-major: BsT rows are contiguous in global -> async b128
      int gn = n0 + ar;
      cp16_to_lds(&BsT[ar][ac * 8], B + (size_t)gn * K + k0 + ac * 8, gn < Nf);
    } else {
      // B[K][Nf] (small weight matrices, L2-hot): scalar transpose stage
      for (int i = tid; i < BN * BK; i += 256) {
        int n = i >> 5, k = i & 31;
        int gn = n0 + n;
        __bf16 v = (__bf16)0.0f;
        if (gn < Nf) v = B[(size_t)(k0 + k) * Nf + gn];
        BsT[n][k] = v;
      }
    }
    wait_lds_ready();       // s_wait_asynccnt 0 (own wave's async copies)
    __syncthreads();        // cross-wave visibility (ds + barrier)

    // -- fragments: per lane, contiguous 16B LDS chunks (k = 16*half + s, same for A and B) --
    v8bf alo = *(const v8bf*)&As[wm * 16 + ml][half * 16];
    v8bf ahi = *(const v8bf*)&As[wm * 16 + ml][half * 16 + 8];
    v16bf a = __builtin_shufflevector(alo, ahi, 0, 1, 2, 3, 4, 5, 6, 7,
                                      8, 9, 10, 11, 12, 13, 14, 15);

    v8bf b0lo = *(const v8bf*)&BsT[wn * 32 + ml][half * 16];
    v8bf b0hi = *(const v8bf*)&BsT[wn * 32 + ml][half * 16 + 8];
    v16bf b0 = __builtin_shufflevector(b0lo, b0hi, 0, 1, 2, 3, 4, 5, 6, 7,
                                       8, 9, 10, 11, 12, 13, 14, 15);

    v8bf b1lo = *(const v8bf*)&BsT[wn * 32 + 16 + ml][half * 16];
    v8bf b1hi = *(const v8bf*)&BsT[wn * 32 + 16 + ml][half * 16 + 8];
    v16bf b1 = __builtin_shufflevector(b1lo, b1hi, 0, 1, 2, 3, 4, 5, 6, 7,
                                       8, 9, 10, 11, 12, 13, 14, 15);

    acc0 = __builtin_amdgcn_wmma_f32_16x16x32_bf16(false, a, false, b0,
                                                   (short)0, acc0, false, false);
    acc1 = __builtin_amdgcn_wmma_f32_16x16x32_bf16(false, a, false, b1,
                                                   (short)0, acc1, false, false);
    __syncthreads();
  }

  // -- store C (C/D layout: VGPR r -> row half*8+r, col = lane&15) --
  const int col0 = n0 + wn * 32 + ml;
  const int col1 = col0 + 16;
  #pragma unroll
  for (int r = 0; r < 8; ++r) {
    int row = m0 + wm * 16 + half * 8 + r;
    if (row < M) {
      if (col0 < Nf) C[(size_t)row * Nf + col0] = acc0[r];
      if (col1 < Nf) C[(size_t)row * Nf + col1] = acc1[r];
    }
  }
}

// ---------------------------------------------------------------------------
extern "C" void kernel_launch(void* const* d_in, const int* in_sizes, int n_in,
                              void* d_out, int out_size, void* d_ws, size_t ws_size,
                              hipStream_t stream) {
  const float* x     = (const float*)d_in[0];
  const int*   ei    = (const int*)d_in[1];
  const float* encW1 = (const float*)d_in[2];
  const float* encb1 = (const float*)d_in[3];
  const float* encW2 = (const float*)d_in[4];
  const float* encb2 = (const float*)d_in[5];
  const float* attW1 = (const float*)d_in[6];
  const float* attb1 = (const float*)d_in[7];
  const float* attW2 = (const float*)d_in[8];
  const float* attb2 = (const float*)d_in[9];
  const float* stW   = (const float*)d_in[10];
  const float* stb   = (const float*)d_in[11];

  const int N = NNODES;
  const int E = in_sizes[1] / 2;
  const int* src = ei;
  const int* dst = ei + E;

  // workspace carve-up: f32 region then bf16 region (all offsets 16B-aligned)
  float* ws   = (float*)d_ws;
  float* deg  = ws;
  float* dinv = deg  + N;
  float* nrm  = dinv + N;
  float* xw   = nrm  + E;                        // N*256 GEMM output scratch
  float* h1   = xw   + (size_t)N * INDIM;        // N*64
  float* h    = h1   + (size_t)N * HIDDIM;       // N*64
  float* t    = h    + (size_t)N * HIDDIM;       // N*64
  float* hs   = t    + (size_t)N * HIDDIM;       // N*256 (struct features h_)
  __bf16* ab  = (__bf16*)(hs + (size_t)N * INDIM);  // N*256 bf16 activations
  __bf16* wb  = ab + (size_t)N * INDIM;             // 64*256 bf16 weights
  __bf16* hsb = wb + (size_t)HIDDIM * INDIM;        // N*256 bf16 h_

  float* x_out = (float*)d_out;                  // N*256
  float* adj   = x_out + (size_t)N * INDIM;      // N*N

  auto cdiv = [](long long a, long long b) { return (int)((a + b - 1) / b); };
  const int T = 256;

  // --- gcn_norm ---
  k_zero  <<<cdiv(N, T), T, 0, stream>>>(deg, N);
  k_degree<<<cdiv(E, T), T, 0, stream>>>(src, dst, E, deg);
  k_dinv  <<<cdiv(N, T), T, 0, stream>>>(deg, dinv, N);
  k_norm  <<<cdiv(E, T), T, 0, stream>>>(src, dst, dinv, nrm, E);

  dim3 gW(1, cdiv(N, 64));                       // Nf=64  -> grid.x = 1
  dim3 gW256(cdiv(INDIM, 64), cdiv(N, 64));      // Nf=256 -> grid.x = 4

  // --- encoder layer 1: h1 = relu(agg(x @ W1)) ---
  k_cvt_bf16<<<cdiv((long long)N * INDIM, T), T, 0, stream>>>(ab, x, N * INDIM);
  k_cvt_bf16<<<cdiv(INDIM * HIDDIM, T), T, 0, stream>>>(wb, encW1, INDIM * HIDDIM);
  gemm_wmma_bf16<false><<<gW, T, 0, stream>>>(ab, wb, xw, N, INDIM, HIDDIM);
  k_selfinit<<<cdiv((long long)N * HIDDIM, T), T, 0, stream>>>(h1, xw, dinv, encb1, N, HIDDIM);
  k_scatter <<<cdiv((long long)E * HIDDIM, T), T, 0, stream>>>(h1, xw, src, dst, nrm, E, HIDDIM);
  k_relu    <<<cdiv((long long)N * HIDDIM, T), T, 0, stream>>>(h1, N * HIDDIM);

  // --- encoder layer 2: h = agg(h1 @ W2) ---
  k_cvt_bf16<<<cdiv((long long)N * HIDDIM, T), T, 0, stream>>>(ab, h1, N * HIDDIM);
  k_cvt_bf16<<<cdiv(HIDDIM * HIDDIM, T), T, 0, stream>>>(wb, encW2, HIDDIM * HIDDIM);
  gemm_wmma_bf16<false><<<gW, T, 0, stream>>>(ab, wb, xw, N, HIDDIM, HIDDIM);
  k_selfinit<<<cdiv((long long)N * HIDDIM, T), T, 0, stream>>>(h, xw, dinv, encb2, N, HIDDIM);
  k_scatter <<<cdiv((long long)E * HIDDIM, T), T, 0, stream>>>(h, xw, src, dst, nrm, E, HIDDIM);

  // --- attr decoder layer 1: t = relu(agg(h @ aW1)) ---
  k_cvt_bf16<<<cdiv((long long)N * HIDDIM, T), T, 0, stream>>>(ab, h, N * HIDDIM);
  k_cvt_bf16<<<cdiv(HIDDIM * HIDDIM, T), T, 0, stream>>>(wb, attW1, HIDDIM * HIDDIM);
  gemm_wmma_bf16<false><<<gW, T, 0, stream>>>(ab, wb, xw, N, HIDDIM, HIDDIM);
  k_selfinit<<<cdiv((long long)N * HIDDIM, T), T, 0, stream>>>(t, xw, dinv, attb1, N, HIDDIM);
  k_scatter <<<cdiv((long long)E * HIDDIM, T), T, 0, stream>>>(t, xw, src, dst, nrm, E, HIDDIM);
  k_relu    <<<cdiv((long long)N * HIDDIM, T), T, 0, stream>>>(t, N * HIDDIM);

  // --- attr decoder layer 2: x_ = agg(t @ aW2) -> d_out[0 : N*256] ---
  k_cvt_bf16<<<cdiv((long long)N * HIDDIM, T), T, 0, stream>>>(ab, t, N * HIDDIM);
  k_cvt_bf16<<<cdiv(HIDDIM * INDIM, T), T, 0, stream>>>(wb, attW2, HIDDIM * INDIM);
  gemm_wmma_bf16<false><<<gW256, T, 0, stream>>>(ab, wb, xw, N, HIDDIM, INDIM);
  k_selfinit<<<cdiv((long long)N * INDIM, T), T, 0, stream>>>(x_out, xw, dinv, attb2, N, INDIM);
  k_scatter <<<cdiv((long long)E * INDIM, T), T, 0, stream>>>(x_out, xw, src, dst, nrm, E, INDIM);

  // --- struct decoder: h_ = agg(h @ sW) ---
  k_cvt_bf16<<<cdiv((long long)N * HIDDIM, T), T, 0, stream>>>(ab, h, N * HIDDIM);
  k_cvt_bf16<<<cdiv(HIDDIM * INDIM, T), T, 0, stream>>>(wb, stW, HIDDIM * INDIM);
  gemm_wmma_bf16<false><<<gW256, T, 0, stream>>>(ab, wb, xw, N, HIDDIM, INDIM);
  k_selfinit<<<cdiv((long long)N * INDIM, T), T, 0, stream>>>(hs, xw, dinv, stb, N, INDIM);
  k_scatter <<<cdiv((long long)E * INDIM, T), T, 0, stream>>>(hs, xw, src, dst, nrm, E, INDIM);

  // --- adj_ = h_ @ h_^T (51.2 GFLOP hot spot) -> d_out[N*256 : ] ---
  k_cvt_bf16<<<cdiv((long long)N * INDIM, T), T, 0, stream>>>(hsb, hs, N * INDIM);
  dim3 gAdj(cdiv(N, 64), cdiv(N, 64));
  gemm_wmma_bf16<true><<<gAdj, T, 0, stream>>>(hsb, hsb, adj, N, INDIM, N);
}
